// FallbackMambaBlock_82257213653611
// MI455X (gfx1250) — compile-verified, matrix operand order checked by reading, not run on hardware
//
#include <hip/hip_runtime.h>
#include <hip/hip_bf16.h>
#include <math.h>

// ---- problem dims (from reference) ----
#define D_MODEL 768
#define D_STATE 64
#define D_CONVW 4
#define D_INNER 1536
#define BATCH   2
#define SEQ     1024
#define ROWS    (BATCH * SEQ)   // 2048
#define XZ_LD   (2 * D_INNER)   // 3072
#define XDBL_COLS 129           // [dt | B(64) | C(64)]
#define XDBL_LD   144           // padded to multiple of 16 for WMMA stores

typedef __attribute__((ext_vector_type(16))) _Float16 v16h;
typedef __attribute__((ext_vector_type(8)))  _Float16 v8h;
typedef __attribute__((ext_vector_type(8)))  float    v8f;

__device__ __forceinline__ float silu_f(float v)     { return v / (1.f + __expf(-v)); }
__device__ __forceinline__ float softplus_f(float v) { return v > 20.f ? v : log1pf(__expf(v)); }

// Load one f16 WMMA operand fragment (no conversion — operands pre-cast to f16).
// ISA 7.12.2 (16-bit A 16x32): lane holds row m=lane&15 (A) / col n=lane&15 (B);
// halves j=0..7 -> K = kb+j, halves j=8..15 -> K = 16+kb+(j-8), kb = (lane>=16)?8:0.
// `p` = row_base + k0 + kb (16B aligned): two contiguous 8-half chunks at +0, +16.
__device__ __forceinline__ v16h load_frag(const _Float16* __restrict__ p) {
  v8h lo = *reinterpret_cast<const v8h*>(p);
  v8h hi = *reinterpret_cast<const v8h*>(p + 16);
  return __builtin_shufflevector(lo, hi, 0, 1, 2, 3, 4, 5, 6, 7,
                                 8, 9, 10, 11, 12, 13, 14, 15);
}

// ---------------------------------------------------------------------------
// Wide GEMM: C[M x N] = A[M x K] * W[N x K]^T (f16 operands, f32 accumulate).
// One wave per 32x64 output tile: 2 A-frags x 4 B-frags -> 8 v_wmma per K step.
// K-loop kept at unroll 1: one live fragment set (48 VGPR) + 64 accumulator
// VGPRs fits without spills; launch_bounds relaxes the occupancy register cap.
// M % 32 == 0, N % 64 == 0, K % 32 == 0 required.
// ---------------------------------------------------------------------------
__global__ void __launch_bounds__(256, 1)
wmma_gemm_h_wide(const _Float16* __restrict__ A, const _Float16* __restrict__ W,
                 float* __restrict__ C, int Mtiles32, int Ntiles64,
                 int K, int ldc) {
  int wave = (int)((blockIdx.x * blockDim.x + threadIdx.x) >> 5);
  if (wave >= Mtiles32 * Ntiles64) return;        // wave-uniform guard
  int lane = threadIdx.x & 31;
  int mt = wave / Ntiles64, nt = wave % Ntiles64;
  int row0 = mt << 5, n0 = nt << 6;

  int mn = lane & 15;
  int kb = (lane >> 4) << 3;                      // 0 or 8

  const _Float16* arow0 = A + (size_t)(row0 + mn) * K + kb;
  const _Float16* arow1 = A + (size_t)(row0 + 16 + mn) * K + kb;
  const _Float16* wrow0 = W + (size_t)(n0 + mn) * K + kb;
  const _Float16* wrow1 = W + (size_t)(n0 + 16 + mn) * K + kb;
  const _Float16* wrow2 = W + (size_t)(n0 + 32 + mn) * K + kb;
  const _Float16* wrow3 = W + (size_t)(n0 + 48 + mn) * K + kb;

  v8f c00 = {}, c01 = {}, c02 = {}, c03 = {};
  v8f c10 = {}, c11 = {}, c12 = {}, c13 = {};

#pragma unroll 1
  for (int k0 = 0; k0 < K; k0 += 32) {
    if (k0 + 32 < K) {
      __builtin_prefetch(arow0 + k0 + 32, 0, 1);  // global_prefetch_b8
      __builtin_prefetch(wrow0 + k0 + 32, 0, 1);
      __builtin_prefetch(wrow2 + k0 + 32, 0, 1);
    }
    v16h a0 = load_frag(arow0 + k0);
    v16h a1 = load_frag(arow1 + k0);
    v16h b0 = load_frag(wrow0 + k0);
    v16h b1 = load_frag(wrow1 + k0);
    v16h b2 = load_frag(wrow2 + k0);
    v16h b3 = load_frag(wrow3 + k0);
    c00 = __builtin_amdgcn_wmma_f32_16x16x32_f16(false, a0, false, b0, (short)0, c00, false, false);
    c01 = __builtin_amdgcn_wmma_f32_16x16x32_f16(false, a0, false, b1, (short)0, c01, false, false);
    c02 = __builtin_amdgcn_wmma_f32_16x16x32_f16(false, a0, false, b2, (short)0, c02, false, false);
    c03 = __builtin_amdgcn_wmma_f32_16x16x32_f16(false, a0, false, b3, (short)0, c03, false, false);
    c10 = __builtin_amdgcn_wmma_f32_16x16x32_f16(false, a1, false, b0, (short)0, c10, false, false);
    c11 = __builtin_amdgcn_wmma_f32_16x16x32_f16(false, a1, false, b1, (short)0, c11, false, false);
    c12 = __builtin_amdgcn_wmma_f32_16x16x32_f16(false, a1, false, b2, (short)0, c12, false, false);
    c13 = __builtin_amdgcn_wmma_f32_16x16x32_f16(false, a1, false, b3, (short)0, c13, false, false);
  }

  // C/D layout: VGPR r -> m = r + 8*(lane>=16), n = lane&15
  int mhi = (lane >> 4) << 3;
  float* crow0 = C + (size_t)(row0 + mhi) * ldc + n0 + mn;
  float* crow1 = C + (size_t)(row0 + 16 + mhi) * ldc + n0 + mn;
#pragma unroll
  for (int r = 0; r < 8; ++r) {
    crow0[(size_t)r * ldc +  0] = c00[r];
    crow0[(size_t)r * ldc + 16] = c01[r];
    crow0[(size_t)r * ldc + 32] = c02[r];
    crow0[(size_t)r * ldc + 48] = c03[r];
    crow1[(size_t)r * ldc +  0] = c10[r];
    crow1[(size_t)r * ldc + 16] = c11[r];
    crow1[(size_t)r * ldc + 32] = c12[r];
    crow1[(size_t)r * ldc + 48] = c13[r];
  }
}

// ---------------------------------------------------------------------------
// Narrow GEMM (16x16 tiles) for the x_dbl projection. W is pre-padded with
// zero rows to 144, so no masking needed and EXEC stays all-1s trivially.
// ---------------------------------------------------------------------------
__global__ void wmma_gemm_h_16(const _Float16* __restrict__ A, const _Float16* __restrict__ W,
                               float* __restrict__ C, int Mtiles, int Ntiles, int K,
                               int ldc) {
  int wave = (int)((blockIdx.x * blockDim.x + threadIdx.x) >> 5);
  if (wave >= Mtiles * Ntiles) return;            // wave-uniform guard
  int lane = threadIdx.x & 31;
  int mt = wave / Ntiles, nt = wave % Ntiles;
  int row0 = mt << 4, n0 = nt << 4;

  int mn = lane & 15;
  int kb = (lane >> 4) << 3;

  const _Float16* arow = A + (size_t)(row0 + mn) * K + kb;
  const _Float16* wrow = W + (size_t)(n0 + mn) * K + kb;

  v8f c = {};
#pragma unroll 2
  for (int k0 = 0; k0 < K; k0 += 32) {
    v16h af = load_frag(arow + k0);
    v16h bf = load_frag(wrow + k0);
    c = __builtin_amdgcn_wmma_f32_16x16x32_f16(false, af, false, bf, (short)0, c, false, false);
  }

  int mhi = (lane >> 4) << 3;
#pragma unroll
  for (int r = 0; r < 8; ++r) {
    C[(size_t)(row0 + r + mhi) * ldc + n0 + mn] = c[r];
  }
}

// Elementwise f32 -> f16 with tail zero-pad (pad rows are trailing in memory).
__global__ void cvt_f16_pad(const float* __restrict__ in, _Float16* __restrict__ out,
                            int valid, int total) {
  int i = blockIdx.x * blockDim.x + threadIdx.x;
  if (i >= total) return;
  out[i] = (i < valid) ? (_Float16)in[i] : (_Float16)0.f;
}

// Depthwise causal conv (width 4) + SiLU; writes f32 (for scan) + f16 (for GEMM).
// Also applies SiLU to the z half of xz.
__global__ void conv_silu_kernel(const float* __restrict__ xz,
                                 const float* __restrict__ conv_w,
                                 const float* __restrict__ conv_b,
                                 float* __restrict__ x_conv,
                                 _Float16* __restrict__ x_conv_h,
                                 float* __restrict__ zs) {
  int idx = blockIdx.x * blockDim.x + threadIdx.x;
  if (idx >= ROWS * D_INNER) return;
  int d = idx % D_INNER;
  int r = idx / D_INNER;
  int l = r % SEQ;
  int b = r / SEQ;

  float acc = conv_b[d];
#pragma unroll
  for (int i = 0; i < D_CONVW; ++i) {
    int li = l - (D_CONVW - 1) + i;
    if (li >= 0) acc += xz[(size_t)(b * SEQ + li) * XZ_LD + d] * conv_w[d * D_CONVW + i];
  }
  float xc = silu_f(acc);
  x_conv[idx] = xc;
  x_conv_h[idx] = (_Float16)xc;
  zs[idx] = silu_f(xz[(size_t)r * XZ_LD + D_INNER + d]);
}

// Selective scan: one thread per (batch, channel). 64-state recurrence in VGPRs.
// Output written directly in f16 (feeds only the W_out WMMA GEMM).
__global__ void mamba_scan_kernel(const float* __restrict__ x_conv,
                                  const float* __restrict__ x_dbl,
                                  const float* __restrict__ zs,
                                  const float* __restrict__ dt_w,
                                  const float* __restrict__ dt_b,
                                  const float* __restrict__ A_log,
                                  const float* __restrict__ D_param,
                                  _Float16* __restrict__ y) {
  __shared__ float sA[D_STATE];
  if (threadIdx.x < D_STATE) sA[threadIdx.x] = -__expf(A_log[threadIdx.x]);
  __syncthreads();

  int idx = blockIdx.x * blockDim.x + threadIdx.x;
  if (idx >= BATCH * D_INNER) return;
  int b = idx / D_INNER;
  int d = idx % D_INNER;

  float dtw = dt_w[d], dtb = dt_b[d], Dp = D_param[d];

  float h[D_STATE];
#pragma unroll
  for (int n = 0; n < D_STATE; ++n) h[n] = 0.f;

  for (int t = 0; t < SEQ; ++t) {
    size_t r = (size_t)(b * SEQ + t);
    float xc = x_conv[r * D_INNER + d];
    const float* xd = x_dbl + r * XDBL_LD;
    float dt = softplus_f(xd[0] * dtw + dtb);
    const float* Bt = xd + 1;
    const float* Ct = xd + 1 + D_STATE;
    float dtxc = dt * xc;
    float acc = 0.f;
#pragma unroll
    for (int n = 0; n < D_STATE; ++n) {
      float dA = __expf(dt * sA[n]);
      float hn = h[n] * dA + dtxc * Bt[n];
      h[n] = hn;
      acc = fmaf(hn, Ct[n], acc);
    }
    float yy = fmaf(Dp, xc, acc);
    y[r * D_INNER + d] = (_Float16)(yy * zs[r * D_INNER + d]);
  }
}

extern "C" void kernel_launch(void* const* d_in, const int* in_sizes, int n_in,
                              void* d_out, int out_size, void* d_ws, size_t ws_size,
                              hipStream_t stream) {
  const float* x       = (const float*)d_in[0];
  const float* W_in    = (const float*)d_in[1];
  const float* conv_w  = (const float*)d_in[2];
  const float* conv_b  = (const float*)d_in[3];
  const float* W_x     = (const float*)d_in[4];
  const float* dt_w    = (const float*)d_in[5];
  const float* dt_b    = (const float*)d_in[6];
  const float* A_log   = (const float*)d_in[7];
  const float* D_param = (const float*)d_in[8];
  const float* W_out   = (const float*)d_in[9];
  float* out = (float*)d_out;

  // ---- workspace layout ----
  char* ws = (char*)d_ws;
  size_t off = 0;
  auto alloc = [&](size_t bytes) { char* p = ws + off; off += (bytes + 255) & ~(size_t)255; return p; };

  float*     xz       = (float*)    alloc((size_t)ROWS * XZ_LD * 4);        // 2048 x 3072 f32
  float*     x_conv   = (float*)    alloc((size_t)ROWS * D_INNER * 4);      // 2048 x 1536 f32
  float*     zs       = (float*)    alloc((size_t)ROWS * D_INNER * 4);      // 2048 x 1536 f32
  float*     x_dbl    = (float*)    alloc((size_t)ROWS * XDBL_LD * 4);      // 2048 x 144 f32
  _Float16*  xh       = (_Float16*) alloc((size_t)ROWS * D_MODEL * 2);      // 2048 x 768 f16
  _Float16*  w_in_h   = (_Float16*) alloc((size_t)XZ_LD * D_MODEL * 2);     // 3072 x 768 f16
  _Float16*  w_x_h    = (_Float16*) alloc((size_t)XDBL_LD * D_INNER * 2);   // 144 x 1536 f16 (zero-padded)
  _Float16*  w_out_h  = (_Float16*) alloc((size_t)D_MODEL * D_INNER * 2);   // 768 x 1536 f16
  _Float16*  x_conv_h = (_Float16*) alloc((size_t)ROWS * D_INNER * 2);      // 2048 x 1536 f16
  _Float16*  yb_h     = (_Float16*) alloc((size_t)ROWS * D_INNER * 2);      // 2048 x 1536 f16

  // 0) one-time f32 -> f16 casts of GEMM operands (W_x zero-padded to 144 rows)
  {
    int n = ROWS * D_MODEL;
    cvt_f16_pad<<<(n + 255) / 256, 256, 0, stream>>>(x, xh, n, n);
  }
  {
    int n = XZ_LD * D_MODEL;
    cvt_f16_pad<<<(n + 255) / 256, 256, 0, stream>>>(W_in, w_in_h, n, n);
  }
  {
    int valid = XDBL_COLS * D_INNER, total = XDBL_LD * D_INNER;
    cvt_f16_pad<<<(total + 255) / 256, 256, 0, stream>>>(W_x, w_x_h, valid, total);
  }
  {
    int n = D_MODEL * D_INNER;
    cvt_f16_pad<<<(n + 255) / 256, 256, 0, stream>>>(W_out, w_out_h, n, n);
  }

  // 1) xz = x @ W_in^T  (2048 x 3072, K=768): 64 x 48 = 3072 waves of 32x64 tiles
  {
    int waves = (ROWS / 32) * (XZ_LD / 64);
    int thr = waves * 32;
    wmma_gemm_h_wide<<<(thr + 255) / 256, 256, 0, stream>>>(
        xh, w_in_h, xz, ROWS / 32, XZ_LD / 64, D_MODEL, XZ_LD);
  }

  // 2) depthwise conv + SiLU (f32 + f16 copies), and SiLU(z)
  {
    int total = ROWS * D_INNER;
    conv_silu_kernel<<<(total + 255) / 256, 256, 0, stream>>>(
        xz, conv_w, conv_b, x_conv, x_conv_h, zs);
  }

  // 3) x_dbl = x_conv @ W_x^T  (2048 x 144 padded, K=1536)
  {
    int waves = (ROWS / 16) * (XDBL_LD / 16);
    int thr = waves * 32;
    wmma_gemm_h_16<<<(thr + 255) / 256, 256, 0, stream>>>(
        x_conv_h, w_x_h, x_dbl, ROWS / 16, XDBL_LD / 16, D_INNER, XDBL_LD);
  }

  // 4) selective scan -> yb = scan(...) * silu(z), stored f16
  {
    int total = BATCH * D_INNER;  // 3072
    mamba_scan_kernel<<<(total + 255) / 256, 256, 0, stream>>>(
        x_conv, x_dbl, zs, dt_w, dt_b, A_log, D_param, yb_h);
  }

  // 5) out = yb @ W_out^T  (2048 x 768, K=1536): 64 x 12 = 768 waves of 32x64 tiles
  {
    int waves = (ROWS / 32) * (D_MODEL / 64);
    int thr = waves * 32;
    wmma_gemm_h_wide<<<(thr + 255) / 256, 256, 0, stream>>>(
        yb_h, w_out_h, out, ROWS / 32, D_MODEL / 64, D_INNER, D_MODEL);
  }
}
